// GruEncoder_50757923504470
// MI455X (gfx1250) — matrix operand order repaired and checked by reading, not compiled, not run on hardware
//
#include <hip/hip_runtime.h>
#include <hip/hip_bf16.h>

typedef __bf16 bf16;
typedef __attribute__((ext_vector_type(16))) __bf16 v16bf;
typedef __attribute__((ext_vector_type(8)))  __bf16 v8bf;
typedef __attribute__((ext_vector_type(8)))  float  v8f;

#define B_  32
#define T_  1024
#define D_  1024
#define H_  1024
#define G_  3072   // 3*H

// ---------------------------------------------------------------------------
// WMMA fragment loaders (contiguous-k layouts, not the ISA table order, wave32, v_wmma_f32_16x16x32_bf16)
// A: 16x32 (MxK) bf16, row-major, stride ld.
//   lane l: m = l&15 ; kb = (l<16)?0:16 ; k=[kb,kb+16) contiguous
__device__ __forceinline__ v16bf load_a_frag(const bf16* __restrict__ A,
                                             int lane, int ld, int k0) {
  const int m  = lane & 15;
  const int kb = (lane < 16) ? 0 : 16;
  const bf16* p = A + (size_t)m * ld + k0 + kb;
  v8bf lo = *(const v8bf*)(p);
  v8bf hi = *(const v8bf*)(p + 8);
  return __builtin_shufflevector(lo, hi, 0,1,2,3,4,5,6,7,8,9,10,11,12,13,14,15);
}

// B: 32x16 (KxN) bf16. W stored row-major as W[n][k] (K contiguous), stride ld.
//   lane l: n = l&15 ; kb = (l<16)?0:16 ; k=[kb,kb+16) contiguous
__device__ __forceinline__ v16bf load_b_frag(const bf16* __restrict__ W,
                                             int lane, int ld, int k0) {
  const int n  = lane & 15;
  const int kb = (lane < 16) ? 0 : 16;
  const bf16* p = W + (size_t)n * ld + k0 + kb;
  v8bf lo = *(const v8bf*)(p);
  v8bf hi = *(const v8bf*)(p + 8);
  return __builtin_shufflevector(lo, hi, 0,1,2,3,4,5,6,7,8,9,10,11,12,13,14,15);
}

__device__ __forceinline__ v8f wmma_bf16(v16bf a, v16bf b, v8f c) {
  return __builtin_amdgcn_wmma_f32_16x16x32_bf16(false, a, false, b,
                                                 (short)0, c, false, false);
}

__device__ __forceinline__ float sigmoidf_(float x) {
  return 1.0f / (1.0f + __expf(-x));
}

// ---------------------------------------------------------------------------
// f32 -> bf16 conversion
__global__ void __launch_bounds__(256) k_cvt_bf16(const float* __restrict__ in,
                                                  bf16* __restrict__ out, int n) {
  int i = blockIdx.x * 256 + threadIdx.x;
  if (i < n) out[i] = (bf16)in[i];
}

// zero h state (f32 + both bf16 buffers) and the grid-barrier counter
__global__ void __launch_bounds__(256) k_init_state(float* __restrict__ h,
                                                    bf16* __restrict__ h0,
                                                    bf16* __restrict__ h1,
                                                    unsigned* __restrict__ counter) {
  int i = blockIdx.x * 256 + threadIdx.x;
  if (i < B_ * H_) { h[i] = 0.0f; h0[i] = (bf16)0.0f; h1[i] = (bf16)0.0f; }
  if (i == 0) *counter = 0u;
}

// ---------------------------------------------------------------------------
// Phase 1: xproj = x @ W_ih^T + b_ih   (M=32768, N=3072, K=1024), bf16 WMMA.
// 8 waves/block; wave w -> 32x32 tile: mt=w&3 (M strip), nt=w>>2 (N strip).
// Block covers 128(M) x 64(N). Two k-steps in flight per loop iteration
// (4 fragment buffer sets -> ~2 k-steps of load/WMMA overlap, 8 WMMAs/iter).
// C written with non-temporal stores (384 MB streaming, > L2).
__global__ void __launch_bounds__(256) k_gemm_xproj(const bf16* __restrict__ X,
                                                    const bf16* __restrict__ W,
                                                    const float* __restrict__ bias,
                                                    float* __restrict__ C) {
  const int lane  = threadIdx.x & 31;
  const int wave  = threadIdx.x >> 5;
  const int mbase = blockIdx.y * 128 + (wave & 3) * 32;
  const int nbase = blockIdx.x * 64 + (wave >> 2) * 32;

  const bf16* A0 = X + (size_t)mbase * D_;
  const bf16* A1 = A0 + (size_t)16 * D_;
  const bf16* B0 = W + (size_t)nbase * D_;
  const bf16* B1 = B0 + (size_t)16 * D_;

  v8f acc00 = {}, acc01 = {}, acc10 = {}, acc11 = {};

  // stage s0 = data for k, stage s1 = data for k+32
  v16bf a0_0 = load_a_frag(A0, lane, D_, 0);
  v16bf a1_0 = load_a_frag(A1, lane, D_, 0);
  v16bf b0_0 = load_b_frag(B0, lane, D_, 0);
  v16bf b1_0 = load_b_frag(B1, lane, D_, 0);
  v16bf a0_1 = load_a_frag(A0, lane, D_, 32);
  v16bf a1_1 = load_a_frag(A1, lane, D_, 32);
  v16bf b0_1 = load_b_frag(B0, lane, D_, 32);
  v16bf b1_1 = load_b_frag(B1, lane, D_, 32);

  for (int k = 0; k < D_ - 64; k += 64) {
    // prefetch k+64 while consuming k
    v16bf a0n0 = load_a_frag(A0, lane, D_, k + 64);
    v16bf a1n0 = load_a_frag(A1, lane, D_, k + 64);
    v16bf b0n0 = load_b_frag(B0, lane, D_, k + 64);
    v16bf b1n0 = load_b_frag(B1, lane, D_, k + 64);
    acc00 = wmma_bf16(a0_0, b0_0, acc00);
    acc01 = wmma_bf16(a0_0, b1_0, acc01);
    acc10 = wmma_bf16(a1_0, b0_0, acc10);
    acc11 = wmma_bf16(a1_0, b1_0, acc11);
    // prefetch k+96 while consuming k+32
    v16bf a0n1 = load_a_frag(A0, lane, D_, k + 96);
    v16bf a1n1 = load_a_frag(A1, lane, D_, k + 96);
    v16bf b0n1 = load_b_frag(B0, lane, D_, k + 96);
    v16bf b1n1 = load_b_frag(B1, lane, D_, k + 96);
    acc00 = wmma_bf16(a0_1, b0_1, acc00);
    acc01 = wmma_bf16(a0_1, b1_1, acc01);
    acc10 = wmma_bf16(a1_1, b0_1, acc10);
    acc11 = wmma_bf16(a1_1, b1_1, acc11);
    a0_0 = a0n0; a1_0 = a1n0; b0_0 = b0n0; b1_0 = b1n0;
    a0_1 = a0n1; a1_1 = a1n1; b0_1 = b0n1; b1_1 = b1n1;
  }
  // epilogue: k = D_-64 and D_-32
  acc00 = wmma_bf16(a0_0, b0_0, acc00);
  acc01 = wmma_bf16(a0_0, b1_0, acc01);
  acc10 = wmma_bf16(a1_0, b0_0, acc10);
  acc11 = wmma_bf16(a1_0, b1_0, acc11);
  acc00 = wmma_bf16(a0_1, b0_1, acc00);
  acc01 = wmma_bf16(a0_1, b1_1, acc01);
  acc10 = wmma_bf16(a1_1, b0_1, acc10);
  acc11 = wmma_bf16(a1_1, b1_1, acc11);

  const int mo = (lane < 16) ? 0 : 8;
  const int nl = lane & 15;
  const float bv0 = bias[nbase + nl];
  const float bv1 = bias[nbase + 16 + nl];
#pragma unroll
  for (int v = 0; v < 8; ++v) {
    __builtin_nontemporal_store(acc00[v] + bv0,
        &C[(size_t)(mbase + mo + v) * G_ + nbase + nl]);
    __builtin_nontemporal_store(acc01[v] + bv1,
        &C[(size_t)(mbase + mo + v) * G_ + nbase + 16 + nl]);
    __builtin_nontemporal_store(acc10[v] + bv0,
        &C[(size_t)(mbase + 16 + mo + v) * G_ + nbase + nl]);
    __builtin_nontemporal_store(acc11[v] + bv1,
        &C[(size_t)(mbase + 16 + mo + v) * G_ + nbase + 16 + nl]);
  }
}

// ---------------------------------------------------------------------------
// Phase 2: persistent GRU scan. 64 blocks (one per 16-wide H slice), 256 thr.
// W_hh slice (3 gates x 16 cols x K=1024 bf16 = 96 KB) is preloaded ONCE into
// LDS in fragment-ready layout, then every timestep's B fragments come from
// ds_load_b128 instead of L2. A fragments (h in bf16) are prefetched one
// k-step ahead; two alternating accumulators break the WMMA C-chain.
__global__ void __launch_bounds__(256) k_gru_scan(const bf16* __restrict__ Whh,
                                                  const float* __restrict__ bhh,
                                                  const float* __restrict__ xproj,
                                                  const float* __restrict__ mask,
                                                  float* __restrict__ H,
                                                  bf16* __restrict__ hbf0,
                                                  bf16* __restrict__ hbf1,
                                                  float* __restrict__ out,
                                                  float* __restrict__ last,
                                                  unsigned* __restrict__ counter) {
  // [gate][kstep][lane][16 bf16]: each lane's fragment chunk is 32 contiguous B
  __shared__ bf16  Wlds[3][32][32][16];          // 96 KB
  __shared__ float hp[3][32][16];                // 6 KB
  const int lane  = threadIdx.x & 31;
  const int wave  = threadIdx.x >> 5;
  const int nbase = blockIdx.x * 16;
  const unsigned nblk = gridDim.x;

  // one-time preload of W_hh fragments into LDS (3072 chunks of 32 B)
  for (int c = threadIdx.x; c < 3 * 32 * 32; c += 256) {
    const int g  = c >> 10;
    const int ks = (c >> 5) & 31;
    const int l  = c & 31;
    const int n  = l & 15;
    const int kb = (l < 16) ? 0 : 16;
    const bf16* src = Whh + (size_t)(g * H_ + nbase + n) * H_ + ks * 32 + kb;
    *(v8bf*)&Wlds[g][ks][l][0] = *(const v8bf*)(src);
    *(v8bf*)&Wlds[g][ks][l][8] = *(const v8bf*)(src + 8);
  }
  __syncthreads();

  for (int t = 0; t < T_; ++t) {
    const bf16* hin  = (t & 1) ? hbf1 : hbf0;
    bf16*       hout = (t & 1) ? hbf0 : hbf1;

    if (wave < 6) {
      const int gate = wave >> 1;
      const int mt   = wave & 1;
      const bf16* Arow = hin + (size_t)(mt * 16) * H_;

      v8f accA = {}, accB = {};
      v16bf a = load_a_frag(Arow, lane, H_, 0);
#pragma unroll 4
      for (int ks = 0; ks < 31; ++ks) {
        v16bf an = load_a_frag(Arow, lane, H_, (ks + 1) * 32);   // prefetch
        v8bf blo = *(const v8bf*)&Wlds[gate][ks][lane][0];
        v8bf bhi = *(const v8bf*)&Wlds[gate][ks][lane][8];
        v16bf b  = __builtin_shufflevector(blo, bhi,
                     0,1,2,3,4,5,6,7,8,9,10,11,12,13,14,15);
        if (ks & 1) accB = wmma_bf16(a, b, accB);
        else        accA = wmma_bf16(a, b, accA);
        a = an;
      }
      {
        v8bf blo = *(const v8bf*)&Wlds[gate][31][lane][0];
        v8bf bhi = *(const v8bf*)&Wlds[gate][31][lane][8];
        v16bf b  = __builtin_shufflevector(blo, bhi,
                     0,1,2,3,4,5,6,7,8,9,10,11,12,13,14,15);
        accB = wmma_bf16(a, b, accB);
      }
      v8f acc = accA + accB;

      const int nl = lane & 15;
      const float bv = bhh[gate * H_ + nbase + nl];
      const int mo = mt * 16 + ((lane < 16) ? 0 : 8);
#pragma unroll
      for (int v = 0; v < 8; ++v) hp[gate][mo + v][nl] = acc[v] + bv;
    }
    __syncthreads();

    // gate fusion: 512 elements (32 rows x 16 cols), 2 per thread
#pragma unroll
    for (int i = 0; i < 2; ++i) {
      const int e = threadIdx.x * 2 + i;
      const int b = e >> 4;
      const int n = e & 15;
      const int col = nbase + n;
      const size_t xrow = ((size_t)b * T_ + t) * G_;
      const float xr = __builtin_nontemporal_load(&xproj[xrow + col]);
      const float xz = __builtin_nontemporal_load(&xproj[xrow + H_ + col]);
      const float xn = __builtin_nontemporal_load(&xproj[xrow + 2 * H_ + col]);
      const float r  = sigmoidf_(xr + hp[0][b][n]);
      const float z  = sigmoidf_(xz + hp[1][b][n]);
      const float nn = tanhf(xn + r * hp[2][b][n]);
      const float hprev = H[b * H_ + col];
      const float m  = mask[b * T_ + t];
      const float hnew = (1.0f - z) * nn + z * hprev;
      const float h2 = m * hnew + (1.0f - m) * hprev;
      H[b * H_ + col] = h2;
      hout[b * H_ + col] = (bf16)h2;
      __builtin_nontemporal_store(h2, &out[((size_t)b * T_ + t) * H_ + col]);
      if (t == T_ - 1) last[b * H_ + col] = h2;
    }

    // grid-wide barrier (monotonic counter; 64 blocks are all co-resident)
    __threadfence();
    __syncthreads();
    if (threadIdx.x == 0) {
      __hip_atomic_fetch_add(counter, 1u, __ATOMIC_RELEASE,
                             __HIP_MEMORY_SCOPE_AGENT);
      const unsigned target = (unsigned)(t + 1) * nblk;
      while (__hip_atomic_load(counter, __ATOMIC_ACQUIRE,
                               __HIP_MEMORY_SCOPE_AGENT) < target) {
        __builtin_amdgcn_s_sleep(2);
      }
    }
    __syncthreads();
  }
}

// ---------------------------------------------------------------------------
extern "C" void kernel_launch(void* const* d_in, const int* in_sizes, int n_in,
                              void* d_out, int out_size, void* d_ws, size_t ws_size,
                              hipStream_t stream) {
  (void)in_sizes; (void)n_in; (void)out_size; (void)ws_size;

  const float* x    = (const float*)d_in[0];   // (B,T,D)
  const float* mask = (const float*)d_in[1];   // (B,T)
  const float* W_ih = (const float*)d_in[2];   // (3H,D)
  const float* W_hh = (const float*)d_in[3];   // (3H,H)
  const float* b_ih = (const float*)d_in[4];   // (3H,)
  const float* b_hh = (const float*)d_in[5];   // (3H,)

  float* last = (float*)d_out;                 // (B,H)
  float* out  = last + (size_t)B_ * H_;        // (B,T,H)

  // workspace carve-up
  char* ws = (char*)d_ws;
  size_t o = 0;
  bf16*  xbf   = (bf16*)(ws + o);  o += (size_t)B_ * T_ * D_ * sizeof(bf16);   // 64 MB
  bf16*  wihbf = (bf16*)(ws + o);  o += (size_t)G_ * D_ * sizeof(bf16);        // 6 MB
  bf16*  whhbf = (bf16*)(ws + o);  o += (size_t)G_ * H_ * sizeof(bf16);        // 6 MB
  float* xproj = (float*)(ws + o); o += (size_t)B_ * T_ * G_ * sizeof(float);  // 384 MB
  float* H     = (float*)(ws + o); o += (size_t)B_ * H_ * sizeof(float);
  bf16*  hbf0  = (bf16*)(ws + o);  o += (size_t)B_ * H_ * sizeof(bf16);
  bf16*  hbf1  = (bf16*)(ws + o);  o += (size_t)B_ * H_ * sizeof(bf16);
  unsigned* counter = (unsigned*)(ws + o);

  // 1) conversions to bf16
  {
    int n = B_ * T_ * D_;
    k_cvt_bf16<<<(n + 255) / 256, 256, 0, stream>>>(x, xbf, n);
  }
  {
    int n = G_ * D_;
    k_cvt_bf16<<<(n + 255) / 256, 256, 0, stream>>>(W_ih, wihbf, n);
    k_cvt_bf16<<<(n + 255) / 256, 256, 0, stream>>>(W_hh, whhbf, n);
  }

  // 2) zero hidden state + barrier counter
  k_init_state<<<(B_ * H_ + 255) / 256, 256, 0, stream>>>(H, hbf0, hbf1, counter);

  // 3) xproj GEMM: grid (N/64, M/128) = (48, 256)
  {
    dim3 grid(G_ / 64, (B_ * T_) / 128);
    k_gemm_xproj<<<grid, 256, 0, stream>>>(xbf, wihbf, b_ih, xproj);
  }

  // 4) persistent scan: 64 blocks (H/16), 256 threads
  k_gru_scan<<<H_ / 16, 256, 0, stream>>>(whhbf, b_hh, xproj, mask,
                                          H, hbf0, hbf1, out, last, counter);
}
